// GNNNicheEncoder_47502338293998
// MI455X (gfx1250) — matrix operand order repaired and checked by reading, not compiled, not run on hardware
//
#include <hip/hip_runtime.h>

#define D 256
#define NN 17
#define BPG 16          // batch elements per workgroup
#define ROWS (BPG*NN)   // 272 = 17 M-tiles of 16
#define HPAD 260        // padded row stride (floats) to stagger LDS banks
#define NWAVES 18       // 17 h-tiles + 1 sum-tile
#define NTHREADS (NWAVES*32)
#define NTILES 16       // 256 / 16
#define KSTEPS 8        // 256 / 32
#define GSZ 2           // B tiles per pipeline group (GSZ=4 spilled; 2 fits)
#define LDS_FLOATS ((ROWS + BPG) * HPAD)

typedef __attribute__((ext_vector_type(16))) _Float16 v16h;
typedef __attribute__((ext_vector_type(8)))  float    v8f;

union AFrag { v16h v; _Float16 h[16]; };
union BFrag { v16h v; uint4 q[2]; };

// ---- prep: build f16 combined weights in workspace ----
// ws layout (f16): [layer][ Wc (65536) | Wl (65536) ]
__global__ void sage_prep_weights(const float* __restrict__ Wl0,
                                  const float* __restrict__ Wr0,
                                  const float* __restrict__ Wl1,
                                  const float* __restrict__ Wr1,
                                  _Float16* __restrict__ ws) {
  int i = blockIdx.x * blockDim.x + threadIdx.x;   // 0..65535
  const float inv16 = 1.0f / 16.0f;
  ws[i]           = (_Float16)(Wr0[i] - Wl0[i] * inv16);
  ws[65536 + i]   = (_Float16)Wl0[i];
  ws[131072 + i]  = (_Float16)(Wr1[i] - Wl1[i] * inv16);
  ws[196608 + i]  = (_Float16)Wl1[i];
}

// ---- fused 2-layer SAGE kernel: 16 batch elements per workgroup ----
extern __shared__ float lds[];   // LDS_FLOATS floats (~293 KB)

__global__ __launch_bounds__(NTHREADS, 1)
void sage_fused(const float* __restrict__ x,
                const float* __restrict__ bl0, const float* __restrict__ g0,
                const float* __restrict__ e0,
                const float* __restrict__ bl1, const float* __restrict__ g1,
                const float* __restrict__ e1,
                const _Float16* __restrict__ wf,
                float* __restrict__ out) {
  const int tid  = threadIdx.x;
  const int wave = tid >> 5;
  const int lane = tid & 31;
  const int half = lane >> 4;
  const int l16  = lane & 15;
  const size_t gbase = (size_t)blockIdx.x * (ROWS * D);

  // ---- coalesced load of the 272x256 f32 h block into LDS ----
  {
    const float4* src = reinterpret_cast<const float4*>(x + gbase);
    for (int i = tid; i < ROWS * D / 4; i += NTHREADS) {
      int e = i * 4;
      int r = e >> 8, c = e & 255;
      *reinterpret_cast<float4*>(&lds[r * HPAD + c]) = src[i];
    }
  }
  __syncthreads();

  for (int layer = 0; layer < 2; ++layer) {
    const _Float16* Wc = wf + layer * 131072;
    const _Float16* Wl = Wc + 65536;
    const float* bl = layer ? bl1 : bl0;
    const float* lg = layer ? g1  : g0;
    const float* lb = layer ? e1  : e0;

    // ---- per-batch node sums s[b][k] into LDS rows [ROWS..ROWS+15] ----
    for (int i = tid; i < BPG * D; i += NTHREADS) {
      int b = i >> 8, k = i & 255;
      float s = 0.f;
      const float* hp = &lds[(b * NN) * HPAD + k];
      #pragma unroll
      for (int n = 0; n < NN; ++n) s += hp[n * HPAD];
      lds[(ROWS + b) * HPAD + k] = s;
    }
    __syncthreads();

    // ---- GEMM: wave w<17 does h-tile w with B=Wc; wave 17 does s-tile with B=Wl ----
    const int arow = (wave < 17) ? (wave * 16 + l16) : (ROWS + l16);
    const _Float16* WB = (wave < 17) ? Wc : Wl;

    v8f acc[NTILES];
    #pragma unroll
    for (int t = 0; t < NTILES; ++t) acc[t] = (v8f){0.f,0.f,0.f,0.f,0.f,0.f,0.f,0.f};

    const _Float16* wbase = WB + (size_t)l16 * D + half * 16;
    const float* rp = &lds[arow * HPAD];

    // kk-major GEMM with grouped, double-buffered B stream.
    // A layout: lane = row M=lane%16; elems 0..15 = K kb+16h.. (16 consecutive K/half);
    // B: lane = col N=lane%16, 16 consecutive K/half.
    #pragma unroll
    for (int kk = 0; kk < KSTEPS; ++kk) {
      AFrag af;
      {
        int kb = kk * 32 + half * 16;
        #pragma unroll
        for (int e = 0; e < 8; ++e) {
          af.h[e]     = (_Float16)rp[kb + e];
          af.h[8 + e] = (_Float16)rp[kb + 8 + e];
        }
      }
      BFrag bf[2][GSZ];
      #pragma unroll
      for (int g = 0; g < GSZ; ++g) {
        const uint4* wq = reinterpret_cast<const uint4*>(
            wbase + (size_t)g * 16 * D + kk * 32);
        bf[0][g].q[0] = wq[0];
        bf[0][g].q[1] = wq[1];
      }
      __builtin_amdgcn_sched_barrier(0);   // pin prologue load group
      #pragma unroll
      for (int tg = 0; tg < NTILES / GSZ; ++tg) {
        int cur = tg & 1;
        if (tg + 1 < NTILES / GSZ) {
          #pragma unroll
          for (int g = 0; g < GSZ; ++g) {
            int t = (tg + 1) * GSZ + g;
            const uint4* wq = reinterpret_cast<const uint4*>(
                wbase + (size_t)t * 16 * D + kk * 32);
            bf[cur ^ 1][g].q[0] = wq[0];
            bf[cur ^ 1][g].q[1] = wq[1];
          }
          __builtin_amdgcn_sched_barrier(0);   // keep next group's loads ahead of WMMAs
        }
        #pragma unroll
        for (int g = 0; g < GSZ; ++g) {
          int t = tg * GSZ + g;
          acc[t] = __builtin_amdgcn_wmma_f32_16x16x32_f16(
              false, af.v, false, bf[cur][g].v, (short)0, acc[t], false, false);
        }
      }
    }

    // ---- wave 17: r[b][o] = (s @ Wl^T)[b][o]/16 + b_l[o], overwrite s rows ----
    if (wave == 17) {
      #pragma unroll
      for (int t = 0; t < NTILES; ++t) {
        int o = t * 16 + l16;
        float blv = bl[o];
        #pragma unroll
        for (int j = 0; j < 8; ++j) {
          int b = j + 8 * half;           // C row M = batch index
          lds[(ROWS + b) * HPAD + o] = acc[t][j] * (1.f / 16.f) + blv;
        }
      }
    }
    __syncthreads();

    // ---- LayerNorm + LeakyReLU + residual, rows owned per-wave ----
    if (wave < 17) {
      float gain[NTILES], bias[NTILES];
      #pragma unroll
      for (int t = 0; t < NTILES; ++t) {
        int o = t * 16 + l16;
        gain[t] = lg[o];
        bias[t] = lb[o];
      }
      #pragma unroll
      for (int j = 0; j < 8; ++j) {
        int m = wave * 16 + j + 8 * half;    // C row
        int b = m / NN;                       // batch of this row
        float v[NTILES];
        float sum = 0.f;
        #pragma unroll
        for (int t = 0; t < NTILES; ++t) {
          v[t] = acc[t][j] + lds[(ROWS + b) * HPAD + t * 16 + l16];
          sum += v[t];
        }
        #pragma unroll
        for (int msk = 1; msk < 16; msk <<= 1) sum += __shfl_xor(sum, msk, 32);
        float mu = sum * (1.f / 256.f);
        float q = 0.f;
        #pragma unroll
        for (int t = 0; t < NTILES; ++t) { float d = v[t] - mu; q += d * d; }
        #pragma unroll
        for (int msk = 1; msk < 16; msk <<= 1) q += __shfl_xor(q, msk, 32);
        float rs = __frsqrt_rn(q * (1.f / 256.f) + 1e-5f);
        #pragma unroll
        for (int t = 0; t < NTILES; ++t) {
          int o = t * 16 + l16;
          float y = (v[t] - mu) * rs * gain[t] + bias[t];
          float a = (y > 0.f) ? y : 0.01f * y;
          lds[m * HPAD + o] = a + lds[m * HPAD + o];   // residual, f32 master in LDS
        }
      }
    }
    __syncthreads();
  }

  // ---- coalesced store ----
  {
    float4* dst = reinterpret_cast<float4*>(out + gbase);
    for (int i = tid; i < ROWS * D / 4; i += NTHREADS) {
      int e = i * 4;
      int r = e >> 8, c = e & 255;
      dst[i] = *reinterpret_cast<const float4*>(&lds[r * HPAD + c]);
    }
  }
}

extern "C" void kernel_launch(void* const* d_in, const int* in_sizes, int n_in,
                              void* d_out, int out_size, void* d_ws, size_t ws_size,
                              hipStream_t stream) {
  const float* x   = (const float*)d_in[0];
  const float* Wl0 = (const float*)d_in[1];
  const float* bl0 = (const float*)d_in[2];
  const float* Wr0 = (const float*)d_in[3];
  const float* g0  = (const float*)d_in[4];
  const float* e0  = (const float*)d_in[5];
  const float* Wl1 = (const float*)d_in[6];
  const float* bl1 = (const float*)d_in[7];
  const float* Wr1 = (const float*)d_in[8];
  const float* g1  = (const float*)d_in[9];
  const float* e1  = (const float*)d_in[10];
  _Float16* wf = (_Float16*)d_ws;

  sage_prep_weights<<<256, 256, 0, stream>>>(Wl0, Wr0, Wl1, Wr1, wf);
  sage_fused<<<16384 / BPG, NTHREADS, LDS_FLOATS * sizeof(float), stream>>>(
      x, bl0, g0, e0, bl1, g1, e1, wf, (float*)d_out);
}
